// GINEModel3_70557722739068
// MI455X (gfx1250) — compile-verified, matrix operand order checked by reading, not run on hardware
//
#include <hip/hip_runtime.h>
#include <hip/hip_bf16.h>
#include <math.h>

// ---------------------------------------------------------------------------
// CDNA5 / gfx1250 fused GINE pipeline, round 2.
// - V_WMMA_F32_16X16X32_F16 for all GEMMs (wave32, f32 accumulate).
// - Edge stage: one wave per 16-edge block; A fragment loaded once, loop over
//   LD/16 column tiles: WMMA + bias + gather h[src] + ReLU + atomic scatter.
//   Edge indices: 1 load/lane + lane-permute (__shfl) broadcast.
// - Node stage: one wave per 16-row block; all K fragments of f16(h+aggr)
//   cached in VGPRs, 16 col tiles x (DIN/32) back-to-back WMMAs.
// - All hot-path index math is 32-bit with compile-time power-of-two strides.
// ---------------------------------------------------------------------------

typedef __attribute__((ext_vector_type(16))) _Float16 v16h;
typedef __attribute__((ext_vector_type(8)))  float    v8f;

#define NNODES 20000
#define NEDGES 320000
#define NGRAPH 512
#define HID    256

// --- WMMA fragment loaders (ISA 7.12.2 layouts) ----------------------------
// 16x32 f16 A/B tile: lane l holds row/col (l&15); K-half (l>>4);
// element e -> K = 8*half + (e<8 ? e : 16+(e-8)).
__device__ __forceinline__ v16h load_frag_f16(const float* __restrict__ rowp, int half) {
    v16h a;
#pragma unroll
    for (int i = 0; i < 8; ++i) {
        a[i]     = (_Float16)rowp[half * 8 + i];
        a[8 + i] = (_Float16)rowp[half * 8 + 16 + i];
    }
    return a;
}

__device__ __forceinline__ v16h load_frag_sum_f16(const float* __restrict__ p0,
                                                  const float* __restrict__ p1, int half) {
    v16h a;
#pragma unroll
    for (int i = 0; i < 8; ++i) {
        int k0 = half * 8 + i;
        int k1 = half * 8 + 16 + i;
        a[i]     = (_Float16)(p0[k0] + p1[k0]);
        a[8 + i] = (_Float16)(p0[k1] + p1[k1]);
    }
    return a;
}

// --- utility ---------------------------------------------------------------
__global__ void zero_f32_kernel(float* __restrict__ p, size_t n) {
    size_t i = (size_t)blockIdx.x * blockDim.x + threadIdx.x;
    size_t stride = (size_t)gridDim.x * blockDim.x;
    for (; i < n; i += stride) p[i] = 0.0f;
}

// --- edge stage ------------------------------------------------------------
// One wave per 16-edge block.  LD = message dim = h stride (64 or 256).
template <int LD>
__global__ void gine_edge_kernel(const float* __restrict__ h,
                                 const float* __restrict__ edge_attr,   // [E,32]
                                 const int*   __restrict__ edge_index,  // [2,E]
                                 const float* __restrict__ we,          // [LD,32]
                                 const float* __restrict__ be,          // [LD]
                                 float* __restrict__ aggr) {            // [N,LD]
    constexpr int NT  = LD / 16;
    constexpr int nEB = NEDGES / 16;
    const int lane = threadIdx.x & 31;
    const int wid  = (blockIdx.x * blockDim.x + threadIdx.x) >> 5;
    if (wid >= nEB) return;

    const int eBase = wid << 4;
    const int half  = lane >> 4;
    const int r     = lane & 15;

    // Speculative prefetch of a future edge block (gfx1250 global_prefetch).
    __builtin_prefetch(edge_attr + (size_t)(eBase + 256) * 32, 0, 1);

    // Edge indices: lanes 0-15 load src, lanes 16-31 load dst; broadcast via
    // lane-permute.  Row j of the D tile belongs to edge eBase+j.
    const int eidx = edge_index[(half ? NEDGES : 0) + eBase + r];
    int sj[8], tj[8];
#pragma unroll
    for (int v = 0; v < 8; ++v) {
        const int j = v + 8 * half;
        sj[v] = __shfl(eidx, j, 32) * LD;         // src row offset (32-bit)
        tj[v] = __shfl(eidx, 16 + j, 32) * LD;    // dst row offset (32-bit)
    }

    // A fragment: 16 edges x 32 edge features, loaded once per wave.
    const v16h a = load_frag_f16(edge_attr + (size_t)(eBase + r) * 32, half);

#pragma unroll
    for (int tile = 0; tile < NT; ++tile) {
        const int n = (tile << 4) + r;            // this lane's channel
        const v16h b = load_frag_f16(we + n * 32, half);
        v8f c = {};
        c = __builtin_amdgcn_wmma_f32_16x16x32_f16(false, a, false, b, (short)0, c,
                                                   false, false);
        const float bias = be[n];
        float hv[8];
#pragma unroll
        for (int v = 0; v < 8; ++v) hv[v] = h[sj[v] + n];   // gather (L2 hit)
#pragma unroll
        for (int v = 0; v < 8; ++v) {
            const float msg = fmaxf(c[v] + bias + hv[v], 0.0f);
            atomicAdd(&aggr[tj[v] + n], msg);
        }
    }
}

// --- node stage: (h + aggr) @ wn^T + bnb, fused BN + ReLU ------------------
// One wave per 16-row block; A fragments for the whole K dim held in VGPRs.
template <int DIN>
__global__ void gine_node_bn_kernel(const float* __restrict__ h,     // [N,DIN]
                                    const float* __restrict__ aggr,  // [N,DIN]
                                    const float* __restrict__ wn,    // [256,DIN]
                                    const float* __restrict__ bnb,
                                    const float* __restrict__ g,
                                    const float* __restrict__ bb,
                                    const float* __restrict__ rm,
                                    const float* __restrict__ rv,
                                    float* __restrict__ out) {       // [N,256]
    constexpr int NK  = DIN / 32;
    constexpr int nRT = NNODES / 16;
    const int lane = threadIdx.x & 31;
    const int wid  = (blockIdx.x * blockDim.x + threadIdx.x) >> 5;
    if (wid >= nRT) return;

    const int rowBase = wid << 4;
    const int half    = lane >> 4;
    const int r       = lane & 15;

    const float* __restrict__ arow0 = h    + (size_t)(rowBase + r) * DIN;
    const float* __restrict__ arow1 = aggr + (size_t)(rowBase + r) * DIN;

    v16h afrag[NK];
#pragma unroll
    for (int k = 0; k < NK; ++k)
        afrag[k] = load_frag_sum_f16(arow0 + k * 32, arow1 + k * 32, half);

    float* __restrict__ orow = out + (size_t)rowBase * HID;

#pragma unroll
    for (int ct = 0; ct < HID / 16; ++ct) {
        const int n = (ct << 4) + r;
        const float* __restrict__ brow = wn + (size_t)n * DIN;
        v8f acc = {};
#pragma unroll
        for (int k = 0; k < NK; ++k) {
            const v16h b = load_frag_f16(brow + k * 32, half);
            acc = __builtin_amdgcn_wmma_f32_16x16x32_f16(false, afrag[k], false, b,
                                                         (short)0, acc, false, false);
        }
        const float scale = g[n] * rsqrtf(rv[n] + 1e-5f);
        const float shift = bb[n] - rm[n] * scale;
        const float bias  = bnb[n];
#pragma unroll
        for (int v = 0; v < 8; ++v) {
            const int m = v + 8 * half;                       // row within tile
            const float val = (acc[v] + bias) * scale + shift;
            orow[m * HID + n] = fmaxf(val, 0.0f);
        }
    }
}

// --- global mean pool ------------------------------------------------------
__global__ void pool_sum_kernel(const float* __restrict__ h,
                                const int* __restrict__ batch,
                                float* __restrict__ sums,  // [G,256]
                                float* __restrict__ cnt) { // [G]
    size_t idx = (size_t)blockIdx.x * blockDim.x + threadIdx.x;
    const size_t total = (size_t)NNODES * HID;
    if (idx >= total) return;
    const int nn = (int)(idx >> 8);
    const int cc = (int)(idx & 255);
    const int gI = batch[nn];
    atomicAdd(&sums[gI * HID + cc], h[idx]);
    if (cc == 0) atomicAdd(&cnt[gI], 1.0f);
}

// --- head: mean, 256->128 ReLU, 128->1 sigmoid -----------------------------
__global__ void head_kernel(const float* __restrict__ sums,
                            const float* __restrict__ cnt,
                            const float* __restrict__ w4,  // [128,256]
                            const float* __restrict__ b4,  // [128]
                            const float* __restrict__ w5,  // [1,128]
                            const float* __restrict__ b5,  // [1]
                            float* __restrict__ outp) {    // [G,1]
    __shared__ float pooled[HID];
    __shared__ float red[128];
    const int gI = blockIdx.x;
    const int t  = threadIdx.x;  // 128 threads

    const float inv = 1.0f / fmaxf(cnt[gI], 1.0f);
    pooled[t]       = sums[gI * HID + t] * inv;
    pooled[t + 128] = sums[gI * HID + 128 + t] * inv;
    __syncthreads();

    float z = b4[t];
    const float* __restrict__ wr = w4 + (size_t)t * HID;
#pragma unroll 4
    for (int k = 0; k < HID; ++k) z = fmaf(pooled[k], wr[k], z);
    z = fmaxf(z, 0.0f);
    red[t] = z * w5[t];
    __syncthreads();
#pragma unroll
    for (int s2 = 64; s2 > 0; s2 >>= 1) {
        if (t < s2) red[t] += red[t + s2];
        __syncthreads();
    }
    if (t == 0) outp[gI] = 1.0f / (1.0f + expf(-(red[0] + b5[0])));
}

// ---------------------------------------------------------------------------
extern "C" void kernel_launch(void* const* d_in, const int* in_sizes, int n_in,
                              void* d_out, int out_size, void* d_ws, size_t ws_size,
                              hipStream_t stream) {
    const float* x    = (const float*)d_in[0];
    const int*   ei   = (const int*)  d_in[1];
    const float* ea   = (const float*)d_in[2];
    const int*   bat  = (const int*)  d_in[3];
    const float* we1  = (const float*)d_in[4];
    const float* be1  = (const float*)d_in[5];
    const float* wn1  = (const float*)d_in[6];
    const float* bnb1 = (const float*)d_in[7];
    const float* we2  = (const float*)d_in[8];
    const float* be2  = (const float*)d_in[9];
    const float* wn2  = (const float*)d_in[10];
    const float* bnb2 = (const float*)d_in[11];
    const float* we3  = (const float*)d_in[12];
    const float* be3  = (const float*)d_in[13];
    const float* wn3  = (const float*)d_in[14];
    const float* bnb3 = (const float*)d_in[15];
    const float* g1 = (const float*)d_in[16], *b1 = (const float*)d_in[17];
    const float* rm1 = (const float*)d_in[18], *rv1 = (const float*)d_in[19];
    const float* g2 = (const float*)d_in[20], *b2 = (const float*)d_in[21];
    const float* rm2 = (const float*)d_in[22], *rv2 = (const float*)d_in[23];
    const float* g3 = (const float*)d_in[24], *b3 = (const float*)d_in[25];
    const float* rm3 = (const float*)d_in[26], *rv3 = (const float*)d_in[27];
    const float* w4 = (const float*)d_in[28], *b4 = (const float*)d_in[29];
    const float* w5 = (const float*)d_in[30], *b5 = (const float*)d_in[31];
    float* outp = (float*)d_out;

    // workspace layout (f32)
    const size_t hBytes = (size_t)NNODES * HID * sizeof(float);       // 20.48 MB
    float* h_a  = (float*)d_ws;                                       // h1 / h3
    float* h_b  = (float*)((char*)d_ws + hBytes);                     // h2
    float* aggr = (float*)((char*)d_ws + 2 * hBytes);                 // [N,256]
    float* sums = (float*)((char*)d_ws + 3 * hBytes);                 // [G,256]
    float* cnt  = (float*)((char*)sums + (size_t)NGRAPH * HID * sizeof(float));
    (void)ws_size; (void)n_in; (void)in_sizes; (void)out_size;

    const int TB = 256;
    auto zero = [&](float* p, size_t n) {
        unsigned blocks = (unsigned)((n + TB - 1) / TB);
        if (blocks > 4096) blocks = 4096;
        zero_f32_kernel<<<blocks, TB, 0, stream>>>(p, n);
    };

    const unsigned edgeBlocks = (NEDGES / 16) * 32 / TB;               // 2500
    const unsigned nodeBlocks = ((NNODES / 16) * 32 + TB - 1) / TB;    // 157

    // ---- layer 1 (din 64) ----
    zero(aggr, (size_t)NNODES * 64);
    gine_edge_kernel<64><<<edgeBlocks, TB, 0, stream>>>(x, ea, ei, we1, be1, aggr);
    gine_node_bn_kernel<64><<<nodeBlocks, TB, 0, stream>>>(
        x, aggr, wn1, bnb1, g1, b1, rm1, rv1, h_a);

    // ---- layer 2 (din 256) ----
    zero(aggr, (size_t)NNODES * HID);
    gine_edge_kernel<256><<<edgeBlocks, TB, 0, stream>>>(h_a, ea, ei, we2, be2, aggr);
    gine_node_bn_kernel<256><<<nodeBlocks, TB, 0, stream>>>(
        h_a, aggr, wn2, bnb2, g2, b2, rm2, rv2, h_b);

    // ---- layer 3 (din 256) ----
    zero(aggr, (size_t)NNODES * HID);
    gine_edge_kernel<256><<<edgeBlocks, TB, 0, stream>>>(h_b, ea, ei, we3, be3, aggr);
    gine_node_bn_kernel<256><<<nodeBlocks, TB, 0, stream>>>(
        h_b, aggr, wn3, bnb3, g3, b3, rm3, rv3, h_a);

    // ---- pool + head ----
    zero(sums, (size_t)NGRAPH * HID);
    zero(cnt, NGRAPH);
    {
        size_t total = (size_t)NNODES * HID;
        pool_sum_kernel<<<(unsigned)((total + TB - 1) / TB), TB, 0, stream>>>(h_a, bat, sums, cnt);
        head_kernel<<<NGRAPH, 128, 0, stream>>>(sums, cnt, w4, b4, w5, b5, outp);
    }
}
